// GraphConvolutionLayer_28157805593140
// MI455X (gfx1250) — matrix operand order repaired and checked by reading, not compiled, not run on hardware
//
#include <hip/hip_runtime.h>

#define N_NODES 16384
#define C_FEAT  128
#define F_FEAT  128

typedef __attribute__((ext_vector_type(16))) __bf16 v16bf;
typedef __attribute__((ext_vector_type(8)))  float  v8f;

// ---------------------------------------------------------------------------
// Kernel 1: d[i] = 1/sqrt(sum_j adj[i,j] + 1)   (self-loop degree, D^{-1/2})
// One 256-thread block per row; coalesced strided read, LDS tree reduce.
// ---------------------------------------------------------------------------
__global__ __launch_bounds__(256)
void gcn_rowsum_kernel(const float* __restrict__ adj, float* __restrict__ dvec) {
    __shared__ float red[256];
    const int row = blockIdx.x;
    const float* rp = adj + (size_t)row * N_NODES;
    float s = 0.0f;
    for (int j = threadIdx.x; j < N_NODES; j += 256) s += rp[j];
    red[threadIdx.x] = s;
    __syncthreads();
    for (int off = 128; off > 0; off >>= 1) {
        if (threadIdx.x < (unsigned)off) red[threadIdx.x] += red[threadIdx.x + off];
        __syncthreads();
    }
    if (threadIdx.x == 0) dvec[row] = 1.0f / sqrtf(red[0] + 1.0f);
}

// ---------------------------------------------------------------------------
// Kernel 2: H = X @ W (tiny: 0.5 GFLOP), Hs = d[n]*H.
// Emits Hs (f32, row-major, for the self-loop term) and HsT (bf16, transposed
// [F][N] so WMMA B-operand loads are contiguous 32B per lane).
// ---------------------------------------------------------------------------
__global__ __launch_bounds__(128)
void gcn_xw_kernel(const float* __restrict__ X, const float* __restrict__ W,
                   const float* __restrict__ dvec,
                   float* __restrict__ Hs, __bf16* __restrict__ HsT) {
    __shared__ float xs[C_FEAT];
    const int n = blockIdx.x;
    const int f = threadIdx.x;
    xs[f] = X[(size_t)n * C_FEAT + f];
    __syncthreads();
    float acc = 0.0f;
#pragma unroll 8
    for (int c = 0; c < C_FEAT; ++c) acc += xs[c] * W[c * F_FEAT + f];
    const float hs = dvec[n] * acc;
    Hs [(size_t)n * F_FEAT  + f] = hs;
    HsT[(size_t)f * N_NODES + n] = (__bf16)hs;
}

// ---------------------------------------------------------------------------
// Kernel 3: out = relu(d[n]*((adj @ Hs)[n,f] + Hs[n,f]) + b[f])
// Block: 256 threads = 8 wave32. Wave w owns rows [blk*128 + w*16, +16),
// all 128 columns as 8 WMMA accumulator tiles (f32). K loop steps by 32:
//   A (16x32): adj fp32 loaded with contiguous k (k = 16h + s), converted to bf16 in-regs.
//   B (32x16): HsT bf16, 16 contiguous values per lane.
// adj streamed from HBM exactly once; HsT reuse hits L2 (4 MiB << 192 MB).
// ---------------------------------------------------------------------------
__global__ __launch_bounds__(256)
void gcn_gemm_kernel(const float* __restrict__ adj,
                     const __bf16* __restrict__ HsT,
                     const float* __restrict__ Hs,
                     const float* __restrict__ dvec,
                     const float* __restrict__ bias,
                     float* __restrict__ out) {
    const int lane = threadIdx.x & 31;
    const int wave = threadIdx.x >> 5;
    const int m0   = blockIdx.x * 128 + wave * 16;
    const int lh   = lane & 15;   // position within 16-lane half
    const int lgrp = lane >> 4;   // 0 or 1

    v8f acc[8] = {};

    // A-operand row for this lane (ISA: A rows striped per lane, M = lane%16)
    const float* arow = adj + (size_t)(m0 + lh) * N_NODES;

    for (int kb = 0; kb < N_NODES; kb += 32) {
        // --- A tile: 16x32 fp32 -> bf16, layout:
        // lane-group 0: halves 0-15 = K kb+0..15
        // lane-group 1: halves 0-15 = K kb+16..31
        const int k1 = kb + lgrp * 16;
        const float4 f0 = *(const float4*)(arow + k1);
        const float4 f1 = *(const float4*)(arow + k1 + 4);
        const float4 f2 = *(const float4*)(arow + k1 + 8);
        const float4 f3 = *(const float4*)(arow + k1 + 12);
        v16bf a;
        a[0]  = (__bf16)f0.x; a[1]  = (__bf16)f0.y; a[2]  = (__bf16)f0.z; a[3]  = (__bf16)f0.w;
        a[4]  = (__bf16)f1.x; a[5]  = (__bf16)f1.y; a[6]  = (__bf16)f1.z; a[7]  = (__bf16)f1.w;
        a[8]  = (__bf16)f2.x; a[9]  = (__bf16)f2.y; a[10] = (__bf16)f2.z; a[11] = (__bf16)f2.w;
        a[12] = (__bf16)f3.x; a[13] = (__bf16)f3.y; a[14] = (__bf16)f3.z; a[15] = (__bf16)f3.w;

        // --- B tiles: 32x16 bf16, lane col N = lane%16, same k order as A:
        // halves h=0..15 hold K = kb + (lane/16)*16 + h  -> contiguous in HsT.
        const int ks = kb + lgrp * 16;
#pragma unroll
        for (int j = 0; j < 8; ++j) {
            const int n = j * 16 + lh;
            const v16bf bmat = *(const v16bf*)(HsT + (size_t)n * N_NODES + ks);
            acc[j] = __builtin_amdgcn_wmma_f32_16x16x32_bf16(
                false, a, false, bmat, (short)0, acc[j], false, false);
        }
    }

    // Epilogue. C/D layout: VGPR r, lane-group g -> M = r + 8*g, N = lane%16.
#pragma unroll
    for (int j = 0; j < 8; ++j) {
        const int col = j * 16 + lh;
        const float bv = bias[col];
#pragma unroll
        for (int r = 0; r < 8; ++r) {
            const int row = m0 + r + lgrp * 8;
            const float v =
                dvec[row] * (acc[j][r] + Hs[(size_t)row * F_FEAT + col]) + bv;
            out[(size_t)row * F_FEAT + col] = fmaxf(v, 0.0f);
        }
    }
}

// ---------------------------------------------------------------------------
extern "C" void kernel_launch(void* const* d_in, const int* in_sizes, int n_in,
                              void* d_out, int out_size, void* d_ws, size_t ws_size,
                              hipStream_t stream) {
    const float* X   = (const float*)d_in[0];
    const float* adj = (const float*)d_in[1];
    const float* W   = (const float*)d_in[2];
    const float* b   = (const float*)d_in[3];
    float* out = (float*)d_out;

    // Workspace layout: d (64 KiB) | HsT bf16 (4 MiB) | Hs f32 (8 MiB)
    float*  dvec = (float*)d_ws;
    __bf16* HsT  = (__bf16*)(dvec + N_NODES);
    float*  Hs   = (float*)((char*)HsT + (size_t)F_FEAT * N_NODES * sizeof(__bf16));

    gcn_rowsum_kernel<<<N_NODES, 256, 0, stream>>>(adj, dvec);
    gcn_xw_kernel<<<N_NODES, C_FEAT, 0, stream>>>(X, W, dvec, Hs, HsT);
    gcn_gemm_kernel<<<N_NODES / 128, 256, 0, stream>>>(adj, HsT, Hs, dvec, b, out);
}
